// TensorForceNet_61581241090606
// MI455X (gfx1250) — compile-verified, matrix operand order checked by reading, not run on hardware
//
#include <hip/hip_runtime.h>

#define H   128
#define R   32
#define NN  4096
#define NE  65536

typedef __bf16 bf16;
typedef bf16  bf16x8  __attribute__((ext_vector_type(8)));
typedef bf16  bf16x16 __attribute__((ext_vector_type(16)));
typedef float f32x8   __attribute__((ext_vector_type(8)));

__device__ __forceinline__ bf16 f2b(float f) {
  union { float f; unsigned u; } v; v.f = f;
  unsigned short h = (unsigned short)(v.u >> 16);
  return __builtin_bit_cast(bf16, h);
}

__device__ __forceinline__ f32x8 wmma_bf16(bf16x16 a, bf16x16 b, f32x8 c) {
  return __builtin_amdgcn_wmma_f32_16x16x32_bf16(false, a, false, b, (short)0, c,
                                                 false, false);
}

// 16x32 bf16 A/B fragment from LDS, row-major [row][k], ld in elements.
// CDNA5 16-bit A layout: lanes 0-15 row=lane K=0..7 & 16..23,
// lanes 16-31 same rows, K=8..15 & 24..31  -> two b128 LDS loads.
__device__ __forceinline__ bf16x16 frag_lds(const bf16* base, int ld, int lane, int k0) {
  int row   = lane & 15;
  int khalf = (lane >> 4) & 1;
  const bf16* p = base + row * ld + k0 + khalf * 8;
  bf16x8 g0 = *(const bf16x8*)(p);
  bf16x8 g1 = *(const bf16x8*)(p + 16);
  return __builtin_shufflevector(g0, g1, 0, 1, 2, 3, 4, 5, 6, 7,
                                 8, 9, 10, 11, 12, 13, 14, 15);
}

// Same fragment from pre-converted bf16 global weights W[row][k] (L2 resident):
// two global_load_b128, no conversion VALU.
__device__ __forceinline__ bf16x16 frag_g(const bf16* W, int ld, int row, int k0,
                                          int lane) {
  int khalf = (lane >> 4) & 1;
  const bf16* p = W + row * ld + k0 + khalf * 8;
  bf16x8 g0 = *(const bf16x8*)(p);
  bf16x8 g1 = *(const bf16x8*)(p + 16);
  return __builtin_shufflevector(g0, g1, 0, 1, 2, 3, 4, 5, 6, 7,
                                 8, 9, 10, 11, 12, 13, 14, 15);
}

__device__ __forceinline__ float siluf(float v) { return v / (1.f + __expf(-v)); }

// ---------------------------------------------------------------------------
// Weight pre-conversion fp32 -> bf16 (runs once per launch, output stays in L2)
// ---------------------------------------------------------------------------
__global__ __launch_bounds__(256) void cvt_kernel(const float* __restrict__ src,
                                                  bf16* __restrict__ dst, int n) {
  const int i = blockIdx.x * 256 + threadIdx.x;
  if (i < n) dst[i] = f2b(src[i]);
}

// ---------------------------------------------------------------------------
// Kernel 1: per-edge geometry + RBF + embedding GEMM + dp GEMMs + scatter-add
// block = 128 threads (4 waves), 16 edges per wave.
// ---------------------------------------------------------------------------
__global__ __launch_bounds__(128) void edge_kernel(
    const float* __restrict__ edge_vec, const bf16* __restrict__ wEmb,
    const bf16* __restrict__ wEmb2, const float* __restrict__ emb2_b,
    const bf16* __restrict__ wDp1, const float* __restrict__ dp1_b,
    const bf16* __restrict__ wDp2, const float* __restrict__ dp2_b,
    const bf16* __restrict__ wDp3, const float* __restrict__ dp3_b,
    const int* __restrict__ z, const int* __restrict__ edge_index,
    float* __restrict__ Iacc, float* __restrict__ Aacc, float* __restrict__ Sacc) {
  __shared__ bf16  sZ[4][16 * 2 * H];   // zpair A tiles (16 x 256)   32 KB
  __shared__ bf16  sRbf[4][16 * R];     // rbf A tiles  (16 x 32)      4 KB
  __shared__ float sGeom[4][16][10];    // cut, n(3), sym(6)
  __shared__ int   sSrc[4][16];

  const int tid = threadIdx.x, wave = tid >> 5, lane = tid & 31;
  const int e0 = (blockIdx.x * 4 + wave) * 16;

  if (lane < 16) {
    const int e = e0 + lane;
    const float vx = edge_vec[3 * e + 0];
    const float vy = edge_vec[3 * e + 1];
    const float vz = edge_vec[3 * e + 2];
    const float d  = sqrtf(vx * vx + vy * vy + vz * vz);
    const float inv = (d > 0.f) ? 1.f / d : 0.f;
    const float nx = vx * inv, ny = vy * inv, nz = vz * inv;
    const float CU = 4.5f;
    const float cut = (d < CU) ? 0.5f * (__cosf(3.14159265358979f * d / CU) + 1.f) : 0.f;
    float* g = &sGeom[wave][lane][0];
    g[0] = cut;
    g[1] = nx; g[2] = ny; g[3] = nz;
    g[4] = nx * nx - (1.f / 3.f); g[5] = nx * ny; g[6] = nx * nz;
    g[7] = ny * ny - (1.f / 3.f); g[8] = ny * nz; g[9] = nz * nz - (1.f / 3.f);
    const int si = edge_index[e];
    const int di = edge_index[NE + e];
    sSrc[wave][lane] = si;
    // expnorm RBF (includes cutoff factor, as in reference)
    const float m0   = __expf(-CU);
    const float dm   = (1.f - m0) / (float)(R - 1);
    const float bden = (2.f / (float)R) * (1.f - m0);
    const float beta = 1.f / (bden * bden);
    const float ex   = __expf(-(5.f / CU) * d);
#pragma unroll
    for (int r = 0; r < R; r++) {
      const float t = ex - (m0 + r * dm);
      sRbf[wave][lane * R + r] = f2b(cut * __expf(-beta * t * t));
    }
    // gather pre-converted bf16 embedding rows with vector copies
    const bf16* es = wEmb + z[si] * H;
    const bf16* ed = wEmb + z[di] * H;
#pragma unroll
    for (int k = 0; k < H; k += 8) {
      *(bf16x8*)&sZ[wave][lane * (2 * H) + k]     = *(const bf16x8*)&es[k];
      *(bf16x8*)&sZ[wave][lane * (2 * H) + H + k] = *(const bf16x8*)&ed[k];
    }
  }
  __syncthreads();

  const int col = lane & 15;
  const bf16x16 ar = frag_lds(&sRbf[wave][0], R, lane, 0);

  for (int nt = 0; nt < 8; nt++) {
    const int n = nt * 16 + col;
    f32x8 accZ = {};
#pragma unroll
    for (int kt = 0; kt < 8; kt++) {
      accZ = wmma_bf16(frag_lds(&sZ[wave][0], 2 * H, lane, kt * 32),
                       frag_g(wEmb2, 2 * H, n, kt * 32, lane), accZ);
    }
    f32x8 a1 = {}, a2 = {}, a3 = {};
    a1 = wmma_bf16(ar, frag_g(wDp1, R, n, 0, lane), a1);
    a2 = wmma_bf16(ar, frag_g(wDp2, R, n, 0, lane), a2);
    a3 = wmma_bf16(ar, frag_g(wDp3, R, n, 0, lane), a3);

    const float bz = emb2_b[n], b1 = dp1_b[n], b2 = dp2_b[n], b3 = dp3_b[n];
#pragma unroll
    for (int r = 0; r < 8; r++) {
      const int m = r + ((lane >> 4) << 3);          // edge within tile
      const float* g = &sGeom[wave][m][0];
      const float c  = g[0] * (accZ[r] + bz);
      const float w1 = (a1[r] + b1) * c;
      const float w2 = (a2[r] + b2) * c;
      const float w3 = (a3[r] + b3) * c;
      const int s = sSrc[wave][m];
      atomicAdd(&Iacc[s * H + n], w1);
      float* ap = &Aacc[(s * H + n) * 3];
      atomicAdd(ap + 0, w2 * g[1]);
      atomicAdd(ap + 1, w2 * g[2]);
      atomicAdd(ap + 2, w2 * g[3]);
      float* sp = &Sacc[(s * H + n) * 6];
#pragma unroll
      for (int j = 0; j < 6; j++) atomicAdd(sp + j, w3 * g[4 + j]);
    }
  }
}

// ---------------------------------------------------------------------------
// Kernel 2: tensor-norm + LayerNorm + ls0/ls1 MLP -> per-channel gates (N,3H)
// block = 128 threads (4 waves), 16 nodes per wave.
// ---------------------------------------------------------------------------
__global__ __launch_bounds__(128) void node_scal_kernel(
    const float* __restrict__ Iacc, const float* __restrict__ Aacc,
    const float* __restrict__ Sacc, const float* __restrict__ init_g,
    const float* __restrict__ init_b, const bf16* __restrict__ wLs0,
    const float* __restrict__ ls0_b, const bf16* __restrict__ wLs1,
    const float* __restrict__ ls1_b, float* __restrict__ scal) {
  __shared__ bf16 sA[4][16 * H];        // normalized tn, bf16        16 KB
  __shared__ bf16 sHl[4][16 * 2 * H];   // silu(ls0) activations      32 KB

  const int tid = threadIdx.x, wave = tid >> 5, lane = tid & 31;
  const int m0 = (blockIdx.x * 4 + wave) * 16;
  const int col = lane & 15;

  if (lane < 16) {
    const int node = m0 + lane;
    float sum = 0.f, sq = 0.f;
    for (int pass = 0; pass < 2; pass++) {
      float mean = 0.f, rstd = 0.f;
      if (pass == 1) {
        mean = sum / (float)H;
        const float var = sq / (float)H - mean * mean;
        rstd = rsqrtf(var + 1e-5f);
      }
      for (int h = 0; h < H; h++) {
        const float iv = Iacc[node * H + h];
        const float* a = &Aacc[(node * H + h) * 3];
        const float* s = &Sacc[(node * H + h) * 6];
        const float tn = 3.f * iv * iv +
                         2.f * (a[0] * a[0] + a[1] * a[1] + a[2] * a[2]) +
                         s[0] * s[0] + s[3] * s[3] + s[5] * s[5] +
                         2.f * (s[1] * s[1] + s[2] * s[2] + s[4] * s[4]);
        if (pass == 0) { sum += tn; sq += tn * tn; }
        else sA[wave][lane * H + h] = f2b((tn - mean) * rstd * init_g[h] + init_b[h]);
      }
    }
  }
  __syncthreads();

  // h = silu(nrm @ ls0_w.T + b)   (16 x 128) @ (128 -> 256)
  for (int nt = 0; nt < 16; nt++) {
    const int n = nt * 16 + col;
    f32x8 acc = {};
#pragma unroll
    for (int kt = 0; kt < 4; kt++)
      acc = wmma_bf16(frag_lds(&sA[wave][0], H, lane, kt * 32),
                      frag_g(wLs0, H, n, kt * 32, lane), acc);
    const float bb = ls0_b[n];
#pragma unroll
    for (int r = 0; r < 8; r++) {
      const int m = r + ((lane >> 4) << 3);
      sHl[wave][m * (2 * H) + n] = f2b(siluf(acc[r] + bb));
    }
  }
  __syncthreads();

  // scal = silu(h @ ls1_w.T + b)   (16 x 256) @ (256 -> 384)
  for (int nt = 0; nt < 24; nt++) {
    const int n = nt * 16 + col;
    f32x8 acc = {};
#pragma unroll
    for (int kt = 0; kt < 8; kt++)
      acc = wmma_bf16(frag_lds(&sHl[wave][0], 2 * H, lane, kt * 32),
                      frag_g(wLs1, 2 * H, n, kt * 32, lane), acc);
    const float bb = ls1_b[n];
#pragma unroll
    for (int r = 0; r < 8; r++) {
      const int m = r + ((lane >> 4) << 3);
      scal[(m0 + m) * (3 * H) + n] = siluf(acc[r] + bb);
    }
  }
}

// ---------------------------------------------------------------------------
// Kernel 3: lt0/lt1/lt2 channel mixing of the 10 tensor components + gated
// squared Frobenius norms -> x (N, 3H).  block = 128 (4 waves), 16 nodes/wave.
// ---------------------------------------------------------------------------
__global__ __launch_bounds__(128) void node_mix_kernel(
    const float* __restrict__ Iacc, const float* __restrict__ Aacc,
    const float* __restrict__ Sacc, const bf16* __restrict__ wLt0,
    const bf16* __restrict__ wLt1, const bf16* __restrict__ wLt2,
    const float* __restrict__ scal, float* __restrict__ x) {
  __shared__ bf16 sT[4][16 * H];  // one component A tile per wave    16 KB

  const int tid = threadIdx.x, wave = tid >> 5, lane = tid & 31;
  const int m0 = (blockIdx.x * 4 + wave) * 16;
  const int col = lane & 15;

  // ---- group I (1 component, lt0, factor 3, gate c=0) ----
  for (int i = lane; i < 16 * H; i += 32) {
    const int m = i >> 7, h = i & 127;
    sT[wave][i] = f2b(Iacc[(m0 + m) * H + h]);
  }
  __syncthreads();
  for (int nt = 0; nt < 8; nt++) {
    const int n = nt * 16 + col;
    f32x8 acc = {};
#pragma unroll
    for (int kt = 0; kt < 4; kt++)
      acc = wmma_bf16(frag_lds(&sT[wave][0], H, lane, kt * 32),
                      frag_g(wLt0, H, n, kt * 32, lane), acc);
#pragma unroll
    for (int r = 0; r < 8; r++) {
      const int m = r + ((lane >> 4) << 3);
      const int node = m0 + m;
      const float g = scal[node * (3 * H) + n * 3 + 0];
      x[node * (3 * H) + n] = 3.f * g * g * acc[r] * acc[r];
    }
  }

  // ---- group A (3 components, lt1, factor 2, gate c=1) ----
  {
    float sqA[8][8];
#pragma unroll
    for (int a = 0; a < 8; a++)
#pragma unroll
      for (int b = 0; b < 8; b++) sqA[a][b] = 0.f;
    for (int cmp = 0; cmp < 3; cmp++) {
      __syncthreads();
      for (int i = lane; i < 16 * H; i += 32) {
        const int m = i >> 7, h = i & 127;
        sT[wave][i] = f2b(Aacc[((m0 + m) * H + h) * 3 + cmp]);
      }
      __syncthreads();
      for (int nt = 0; nt < 8; nt++) {
        const int n = nt * 16 + col;
        f32x8 acc = {};
#pragma unroll
        for (int kt = 0; kt < 4; kt++)
          acc = wmma_bf16(frag_lds(&sT[wave][0], H, lane, kt * 32),
                          frag_g(wLt1, H, n, kt * 32, lane), acc);
#pragma unroll
        for (int r = 0; r < 8; r++) sqA[nt][r] += acc[r] * acc[r];
      }
    }
    for (int nt = 0; nt < 8; nt++) {
      const int n = nt * 16 + col;
#pragma unroll
      for (int r = 0; r < 8; r++) {
        const int m = r + ((lane >> 4) << 3);
        const int node = m0 + m;
        const float g = scal[node * (3 * H) + n * 3 + 1];
        x[node * (3 * H) + H + n] = 2.f * g * g * sqA[nt][r];
      }
    }
  }

  // ---- group S (6 components, lt2, weights {1,2,2,1,2,1}, gate c=2) ----
  {
    float sqS[8][8];
#pragma unroll
    for (int a = 0; a < 8; a++)
#pragma unroll
      for (int b = 0; b < 8; b++) sqS[a][b] = 0.f;
    const float wS[6] = {1.f, 2.f, 2.f, 1.f, 2.f, 1.f};
    for (int cmp = 0; cmp < 6; cmp++) {
      __syncthreads();
      for (int i = lane; i < 16 * H; i += 32) {
        const int m = i >> 7, h = i & 127;
        sT[wave][i] = f2b(Sacc[((m0 + m) * H + h) * 6 + cmp]);
      }
      __syncthreads();
      for (int nt = 0; nt < 8; nt++) {
        const int n = nt * 16 + col;
        f32x8 acc = {};
#pragma unroll
        for (int kt = 0; kt < 4; kt++)
          acc = wmma_bf16(frag_lds(&sT[wave][0], H, lane, kt * 32),
                          frag_g(wLt2, H, n, kt * 32, lane), acc);
#pragma unroll
        for (int r = 0; r < 8; r++) sqS[nt][r] += wS[cmp] * acc[r] * acc[r];
      }
    }
    for (int nt = 0; nt < 8; nt++) {
      const int n = nt * 16 + col;
#pragma unroll
      for (int r = 0; r < 8; r++) {
        const int m = r + ((lane >> 4) << 3);
        const int node = m0 + m;
        const float g = scal[node * (3 * H) + n * 3 + 2];
        x[node * (3 * H) + 2 * H + n] = g * g * sqS[nt][r];
      }
    }
  }
}

// ---------------------------------------------------------------------------
// Kernel 4: LayerNorm(x) + lin + silu + ol1 + silu + ol2 -> out (N,1)
// block = 64 threads (2 waves), 16 nodes per wave.
// ---------------------------------------------------------------------------
__global__ __launch_bounds__(64) void node_out_kernel(
    const float* __restrict__ x, const float* __restrict__ outn_g,
    const float* __restrict__ outn_b, const bf16* __restrict__ wLin,
    const float* __restrict__ lin_b, const bf16* __restrict__ wOl1,
    const float* __restrict__ ol1_b, const float* __restrict__ ol2_w,
    const float* __restrict__ ol2_b, float* __restrict__ out) {
  __shared__ bf16  sX[2][16 * 3 * H];   // normalized x  (16 x 384)   24 KB
  __shared__ bf16  sS1[2][16 * H];      // silu(s1)       (16 x 128)   8 KB
  __shared__ float sS2[2][16 * 64];     // silu(s2)       (16 x 64)    8 KB

  const int tid = threadIdx.x, wave = tid >> 5, lane = tid & 31;
  const int m0 = (blockIdx.x * 2 + wave) * 16;
  const int col = lane & 15;

  if (lane < 16) {
    const int node = m0 + lane;
    const float* xr = &x[node * (3 * H)];
    float sum = 0.f, sq = 0.f;
    for (int k = 0; k < 3 * H; k++) { const float v = xr[k]; sum += v; sq += v * v; }
    const float mean = sum / (float)(3 * H);
    const float var = sq / (float)(3 * H) - mean * mean;
    const float rstd = rsqrtf(var + 1e-5f);
    for (int k = 0; k < 3 * H; k++)
      sX[wave][lane * (3 * H) + k] = f2b((xr[k] - mean) * rstd * outn_g[k] + outn_b[k]);
  }
  __syncthreads();

  // s1 = silu(xln @ lin_w.T + b)   (16 x 384) @ (384 -> 128)
  for (int nt = 0; nt < 8; nt++) {
    const int n = nt * 16 + col;
    f32x8 acc = {};
#pragma unroll
    for (int kt = 0; kt < 12; kt++)
      acc = wmma_bf16(frag_lds(&sX[wave][0], 3 * H, lane, kt * 32),
                      frag_g(wLin, 3 * H, n, kt * 32, lane), acc);
    const float bb = lin_b[n];
#pragma unroll
    for (int r = 0; r < 8; r++) {
      const int m = r + ((lane >> 4) << 3);
      sS1[wave][m * H + n] = f2b(siluf(acc[r] + bb));
    }
  }
  __syncthreads();

  // s2 = silu(s1 @ ol1_w.T + b)    (16 x 128) @ (128 -> 64)
  for (int nt = 0; nt < 4; nt++) {
    const int n = nt * 16 + col;
    f32x8 acc = {};
#pragma unroll
    for (int kt = 0; kt < 4; kt++)
      acc = wmma_bf16(frag_lds(&sS1[wave][0], H, lane, kt * 32),
                      frag_g(wOl1, H, n, kt * 32, lane), acc);
    const float bb = ol1_b[n];
#pragma unroll
    for (int r = 0; r < 8; r++) {
      const int m = r + ((lane >> 4) << 3);
      sS2[wave][m * 64 + n] = siluf(acc[r] + bb);
    }
  }
  __syncthreads();

  if (lane < 16) {
    const int node = m0 + lane;
    float s = 0.f;
    for (int k = 0; k < 64; k++) s += sS2[wave][lane * 64 + k] * ol2_w[k];
    out[node] = s + ol2_b[0];
  }
}

// ---------------------------------------------------------------------------
extern "C" void kernel_launch(void* const* d_in, const int* in_sizes, int n_in,
                              void* d_out, int out_size, void* d_ws, size_t ws_size,
                              hipStream_t stream) {
  const float* edge_vec = (const float*)d_in[0];
  const float* emb_w    = (const float*)d_in[1];
  const float* emb2_w   = (const float*)d_in[2];
  const float* emb2_b   = (const float*)d_in[3];
  const float* dp1_w    = (const float*)d_in[4];
  const float* dp1_b    = (const float*)d_in[5];
  const float* dp2_w    = (const float*)d_in[6];
  const float* dp2_b    = (const float*)d_in[7];
  const float* dp3_w    = (const float*)d_in[8];
  const float* dp3_b    = (const float*)d_in[9];
  const float* lt0_w    = (const float*)d_in[10];
  const float* lt1_w    = (const float*)d_in[11];
  const float* lt2_w    = (const float*)d_in[12];
  const float* ls0_w    = (const float*)d_in[13];
  const float* ls0_b    = (const float*)d_in[14];
  const float* ls1_w    = (const float*)d_in[15];
  const float* ls1_b    = (const float*)d_in[16];
  const float* init_g   = (const float*)d_in[17];
  const float* init_b   = (const float*)d_in[18];
  const float* lin_w    = (const float*)d_in[19];
  const float* lin_b    = (const float*)d_in[20];
  const float* outn_g   = (const float*)d_in[21];
  const float* outn_b   = (const float*)d_in[22];
  const float* ol1_w    = (const float*)d_in[23];
  const float* ol1_b    = (const float*)d_in[24];
  const float* ol2_w    = (const float*)d_in[25];
  const float* ol2_b    = (const float*)d_in[26];
  const int*   z        = (const int*)d_in[27];
  const int*   edge_idx = (const int*)d_in[28];

  float* ws   = (float*)d_ws;
  float* Iacc = ws;                    // N*H            =  524288
  float* Aacc = ws + 524288;           // N*H*3          = 1572864
  float* Sacc = ws + 2097152;          // N*H*6          = 3145728
  float* scal = ws + 5242880;          // N*3H           = 1572864
  float* xbuf = ws + 6815744;          // N*3H           = 1572864   (end 8388608 f32)

  // bf16 weight mirror region (0.6 MB), after the fp32 region
  bf16* B     = (bf16*)(ws + 8388608);
  bf16* wEmb  = B;                     // 128*128 = 16384
  bf16* wEmb2 = B + 16384;             // 128*256 = 32768
  bf16* wDp1  = B + 49152;             // 128*32
  bf16* wDp2  = B + 53248;
  bf16* wDp3  = B + 57344;
  bf16* wLt0  = B + 61440;             // 128*128
  bf16* wLt1  = B + 77824;
  bf16* wLt2  = B + 94208;
  bf16* wLs0  = B + 110592;            // 256*128 = 32768
  bf16* wLs1  = B + 143360;            // 384*256 = 98304
  bf16* wLin  = B + 241664;            // 128*384 = 49152
  bf16* wOl1  = B + 290816;            // 64*128  = 8192

  // zero the scatter accumulators (I, A, S)
  hipMemsetAsync(d_ws, 0, (size_t)5242880 * sizeof(float), stream);

  // pre-convert weights to bf16 (stream-ordered, graph-capturable)
  auto cvt = [&](const float* s, bf16* d, int n) {
    cvt_kernel<<<(n + 255) / 256, 256, 0, stream>>>(s, d, n);
  };
  cvt(emb_w,  wEmb,  16384);
  cvt(emb2_w, wEmb2, 32768);
  cvt(dp1_w,  wDp1,  4096);
  cvt(dp2_w,  wDp2,  4096);
  cvt(dp3_w,  wDp3,  4096);
  cvt(lt0_w,  wLt0,  16384);
  cvt(lt1_w,  wLt1,  16384);
  cvt(lt2_w,  wLt2,  16384);
  cvt(ls0_w,  wLs0,  32768);
  cvt(ls1_w,  wLs1,  98304);
  cvt(lin_w,  wLin,  49152);
  cvt(ol1_w,  wOl1,  8192);

  edge_kernel<<<NE / 64, 128, 0, stream>>>(edge_vec, wEmb, wEmb2, emb2_b, wDp1,
                                           dp1_b, wDp2, dp2_b, wDp3, dp3_b, z,
                                           edge_idx, Iacc, Aacc, Sacc);
  node_scal_kernel<<<NN / 64, 128, 0, stream>>>(Iacc, Aacc, Sacc, init_g, init_b,
                                                wLs0, ls0_b, wLs1, ls1_b, scal);
  node_mix_kernel<<<NN / 64, 128, 0, stream>>>(Iacc, Aacc, Sacc, wLt0, wLt1,
                                               wLt2, scal, xbuf);
  node_out_kernel<<<NN / 32, 64, 0, stream>>>(xbuf, outn_g, outn_b, wLin, lin_b,
                                              wOl1, ol1_b, ol2_w, ol2_b,
                                              (float*)d_out);
}
